// GroupTorchGRU_19636590477496
// MI455X (gfx1250) — matrix operand order repaired and checked by reading, not compiled, not run on hardware
//
#include <hip/hip_runtime.h>

// Problem dims (fixed by the reference)
#define B_DIM 1024
#define U_DIM 8
#define I_DIM 512
#define H_DIM 512
#define G3    (3 * H_DIM)

#define KSTEP  32    // bf16 WMMA K
#define KITERS (I_DIM / KSTEP)
#define TILE_B 128   // rows per workgroup (8 waves x 16)
#define TILE_N 32    // output cols per workgroup (2 n-subtiles per wave)

typedef __attribute__((ext_vector_type(16))) __bf16        v16bf;
typedef __attribute__((ext_vector_type(8)))  float         v8f;
typedef __attribute__((ext_vector_type(4)))  float         f32x4;
typedef __attribute__((ext_vector_type(4)))  unsigned int  u32x4;
typedef __attribute__((ext_vector_type(2)))  unsigned int  u32x2;

union Frag {
    unsigned short us[16];
    unsigned int   u[8];
    u32x4          q[2];
    v16bf          v;
};

// f32 -> packed bf16x2. Plain casts: backend picks native bf16 cvt if the
// target has it, otherwise lowers to the standard RNE software sequence.
__device__ __forceinline__ unsigned int pack2bf(float a, float b) {
    typedef __bf16 bf16x2_t __attribute__((ext_vector_type(2)));
    bf16x2_t v;
    v.x = (__bf16)a;
    v.y = (__bf16)b;
    union { bf16x2_t v; unsigned int u; } c;
    c.v = v;
    return c.u;
}

__device__ __forceinline__ float fast_rcp(float x) {
    return __builtin_amdgcn_rcpf(x);
}
__device__ __forceinline__ float fast_sigmoid(float x) {
    return fast_rcp(1.0f + __expf(-x));
}
__device__ __forceinline__ float fast_tanh(float x) {
    float e = __expf(2.0f * x);
    return 1.0f - 2.0f * fast_rcp(e + 1.0f);
}

__device__ __forceinline__ void cvt8(unsigned int* dst, f32x4 a, f32x4 b) {
    dst[0] = pack2bf(a.x, a.y);
    dst[1] = pack2bf(a.z, a.w);
    dst[2] = pack2bf(b.x, b.y);
    dst[3] = pack2bf(b.z, b.w);
}

// ---------------- pass 1: streaming f32 -> bf16 conversion ----------------
__global__ __launch_bounds__(256)
void cvt_f32_bf16_kernel(const float* __restrict__ src,
                         unsigned short* __restrict__ dst, int n8) {
    int i = blockIdx.x * blockDim.x + threadIdx.x;
    if (i >= n8) return;
    const f32x4* s = (const f32x4*)src + (size_t)i * 2;
    f32x4 a = s[0];
    f32x4 b = s[1];
    u32x4 o = { pack2bf(a.x, a.y), pack2bf(a.z, a.w),
                pack2bf(b.x, b.y), pack2bf(b.z, b.w) };
    *((u32x4*)dst + (size_t)i) = o;
}

// ---------------- pass 2: bf16 WMMA GEMM + GRU epilogue ----------------
// Stage one 12KB B-tile (12 blocks of 16 rows x 32 k, bf16) into an LDS buffer.
__device__ __forceinline__ void stage_B_bf16(unsigned short* __restrict__ sBbuf,
                                             const unsigned short* __restrict__ wih,
                                             const unsigned short* __restrict__ whh,
                                             int tid, int ncol0, int k) {
    #pragma unroll
    for (int it = 0; it < 3; ++it) {
        int idx = tid + it * 256;      // 0..767 chunks of 8 bf16
        int r   = idx >> 2;            // 0..191
        int q   = idx & 3;
        int blk = r >> 4;
        int row = r & 15;
        int mat  = blk >= 6;
        int rem  = blk - (mat ? 6 : 0);
        int gate = rem >> 1;
        int sub  = rem & 1;
        size_t grow = (size_t)(gate * H_DIM + ncol0 + sub * 16 + row);
        const unsigned short* src = (mat ? whh : wih) + grow * I_DIM + k + q * 8;
        u32x4 v = *(const u32x4*)src;
        *(u32x4*)&sBbuf[r * KSTEP + q * 8] = v;
    }
}

__global__ __launch_bounds__(256)
void gru_wmma_bf16_pre(const unsigned short* __restrict__ wsX,    // [B,U,I] bf16
                       const unsigned short* __restrict__ wsH,    // [B,U,H] bf16
                       const unsigned short* __restrict__ wsWih,  // [U,3H,I] bf16
                       const unsigned short* __restrict__ wsWhh,  // [U,3H,H] bf16
                       const float* __restrict__ hidden,          // [B,U,H] f32
                       const float* __restrict__ bih,
                       const float* __restrict__ bhh,
                       float* __restrict__ out) {
    __shared__ __align__(16) unsigned short sB[2][12 * 16 * KSTEP];   // 2 x 12KB

    const int tid   = threadIdx.x;
    const int wave  = tid >> 5;      // wave32
    const int lane  = tid & 31;
    const int ntile = blockIdx.x;    // 0..15
    const int btile = blockIdx.y;    // 0..7
    const int u     = blockIdx.z;    // 0..7
    const int ncol0 = ntile * TILE_N;

    const unsigned short* wih = wsWih + (size_t)u * G3 * I_DIM;
    const unsigned short* whh = wsWhh + (size_t)u * G3 * H_DIM;

    const int m    = lane & 15;
    const int half = lane >> 4;

    const size_t arow = (size_t)(btile * TILE_B + wave * 16 + m) * (size_t)(U_DIM * I_DIM);
    const unsigned short* pax = wsX + (size_t)u * I_DIM + arow;
    const unsigned short* pah = wsH + (size_t)u * H_DIM + arow;
    const int koffA0 = half * 16;
    const int koffA1 = half * 16 + 8;

    v8f accR[2]  = {v8f{}, v8f{}};
    v8f accZ[2]  = {v8f{}, v8f{}};
    v8f accXn[2] = {v8f{}, v8f{}};
    v8f accHn[2] = {v8f{}, v8f{}};

    stage_B_bf16(&sB[0][0], wih, whh, tid, ncol0, 0);
    __syncthreads();

    for (int ki = 0; ki < KITERS; ++ki) {
        const int k   = ki * KSTEP;
        const int buf = ki & 1;
        const unsigned short* sBb = &sB[buf][0];

        // ---- fragment loads first (ds_loads not queued behind staging) ----
        Frag ax, ah;
        ax.q[0] = *(const u32x4*)(pax + k + koffA0);
        ax.q[1] = *(const u32x4*)(pax + k + koffA1);
        ah.q[0] = *(const u32x4*)(pah + k + koffA0);
        ah.q[1] = *(const u32x4*)(pah + k + koffA1);

        Frag b[12];
        #pragma unroll
        for (int blk = 0; blk < 12; ++blk) {
            const unsigned short* rp = &sBb[(blk * 16 + m) * KSTEP + half * 16];
            b[blk].q[0] = *(const u32x4*)(rp);
            b[blk].q[1] = *(const u32x4*)(rp + 8);
        }

        // ---- overlap: stage next k-tile into the other LDS buffer ----
        if (ki + 1 < KITERS)
            stage_B_bf16(&sB[buf ^ 1][0], wih, whh, tid, ncol0, k + KSTEP);

        // ---- 12 WMMAs; blk = mat*6 + gate*2 + sub ----
        #pragma unroll
        for (int sub = 0; sub < 2; ++sub) {
            accR[sub]  = __builtin_amdgcn_wmma_f32_16x16x32_bf16(false, ax.v, false, b[0 + sub].v,  (short)0, accR[sub],  false, false);
            accZ[sub]  = __builtin_amdgcn_wmma_f32_16x16x32_bf16(false, ax.v, false, b[2 + sub].v,  (short)0, accZ[sub],  false, false);
            accXn[sub] = __builtin_amdgcn_wmma_f32_16x16x32_bf16(false, ax.v, false, b[4 + sub].v,  (short)0, accXn[sub], false, false);
            accR[sub]  = __builtin_amdgcn_wmma_f32_16x16x32_bf16(false, ah.v, false, b[6 + sub].v,  (short)0, accR[sub],  false, false);
            accZ[sub]  = __builtin_amdgcn_wmma_f32_16x16x32_bf16(false, ah.v, false, b[8 + sub].v,  (short)0, accZ[sub],  false, false);
            accHn[sub] = __builtin_amdgcn_wmma_f32_16x16x32_bf16(false, ah.v, false, b[10 + sub].v, (short)0, accHn[sub], false, false);
        }

        __syncthreads();
    }

    // ---- epilogue ----
    const int row0 = btile * TILE_B + wave * 16 + half * 8;
    #pragma unroll
    for (int sub = 0; sub < 2; ++sub) {
        const int   col = ncol0 + sub * 16 + m;
        const float br_ = bih[u * G3 + col]         + bhh[u * G3 + col];
        const float bz_ = bih[u * G3 + H_DIM + col] + bhh[u * G3 + H_DIM + col];
        const float bxn = bih[u * G3 + 2 * H_DIM + col];
        const float bhn = bhh[u * G3 + 2 * H_DIM + col];
        #pragma unroll
        for (int j = 0; j < 8; ++j) {
            const size_t off = (size_t)(row0 + j) * (size_t)(U_DIM * H_DIM)
                             + (size_t)u * H_DIM + (size_t)col;
            const float hp = hidden[off];
            const float r  = fast_sigmoid(accR[sub][j] + br_);
            const float z  = fast_sigmoid(accZ[sub][j] + bz_);
            const float n  = fast_tanh(accXn[sub][j] + bxn + r * (accHn[sub][j] + bhn));
            out[off] = (1.0f - z) * n + z * hp;
        }
    }
}

// ---------------- fallback: fused conversion (no workspace) ----------------
__global__ __launch_bounds__(256)
void gru_wmma_bf16_fused(const float* __restrict__ inputs,
                         const float* __restrict__ hidden,
                         const float* __restrict__ Wih,
                         const float* __restrict__ Whh,
                         const float* __restrict__ bih,
                         const float* __restrict__ bhh,
                         float* __restrict__ out) {
    __shared__ __align__(16) unsigned short sB[12 * 16 * KSTEP];

    const int tid   = threadIdx.x;
    const int wave  = tid >> 5;
    const int lane  = tid & 31;
    const int ntile = blockIdx.x;
    const int btile = blockIdx.y;
    const int u     = blockIdx.z;
    const int ncol0 = ntile * TILE_N;

    const float* gWih = Wih + (size_t)u * G3 * I_DIM;
    const float* gWhh = Whh + (size_t)u * G3 * H_DIM;

    const int m    = lane & 15;
    const int half = lane >> 4;

    const size_t arow = (size_t)(btile * TILE_B + wave * 16 + m) * (size_t)(U_DIM * I_DIM);
    const float* pax = inputs + (size_t)u * I_DIM + arow;
    const float* pah = hidden + (size_t)u * H_DIM + arow;
    const int koffA0 = half * 16;
    const int koffA1 = half * 16 + 8;

    v8f accR[2]  = {v8f{}, v8f{}};
    v8f accZ[2]  = {v8f{}, v8f{}};
    v8f accXn[2] = {v8f{}, v8f{}};
    v8f accHn[2] = {v8f{}, v8f{}};

    for (int k = 0; k < I_DIM; k += KSTEP) {
        __syncthreads();
        #pragma unroll
        for (int it = 0; it < 6; ++it) {
            int idx  = tid + it * 256;
            int r    = idx >> 3;
            int q    = idx & 7;
            int blk  = r >> 4;
            int row  = r & 15;
            int mat  = blk >= 6;
            int rem  = blk - (mat ? 6 : 0);
            int gate = rem >> 1;
            int sub  = rem & 1;
            int grow = gate * H_DIM + ncol0 + sub * 16 + row;
            const float* src = (mat ? gWhh : gWih) + (size_t)grow * I_DIM + k + q * 4;
            f32x4 f = *(const f32x4*)(src);
            u32x2 p = { pack2bf(f.x, f.y), pack2bf(f.z, f.w) };
            *(u32x2*)&sB[r * KSTEP + q * 4] = p;
        }
        __syncthreads();

        Frag ax, ah;
        {
            f32x4 x0 = *(const f32x4*)(pax + k + koffA0);
            f32x4 x1 = *(const f32x4*)(pax + k + koffA0 + 4);
            f32x4 x2 = *(const f32x4*)(pax + k + koffA1);
            f32x4 x3 = *(const f32x4*)(pax + k + koffA1 + 4);
            cvt8(&ax.u[0], x0, x1);
            cvt8(&ax.u[4], x2, x3);
            f32x4 h0 = *(const f32x4*)(pah + k + koffA0);
            f32x4 h1 = *(const f32x4*)(pah + k + koffA0 + 4);
            f32x4 h2 = *(const f32x4*)(pah + k + koffA1);
            f32x4 h3 = *(const f32x4*)(pah + k + koffA1 + 4);
            cvt8(&ah.u[0], h0, h1);
            cvt8(&ah.u[4], h2, h3);
        }

        #pragma unroll
        for (int sub = 0; sub < 2; ++sub) {
            Frag b0, b1, b2;
            const unsigned short* rp;
            rp = &sB[((0 + sub) * 16 + m) * KSTEP + half * 16];
            b0.q[0] = *(const u32x4*)(rp);  b0.q[1] = *(const u32x4*)(rp + 8);
            rp = &sB[((2 + sub) * 16 + m) * KSTEP + half * 16];
            b1.q[0] = *(const u32x4*)(rp);  b1.q[1] = *(const u32x4*)(rp + 8);
            rp = &sB[((4 + sub) * 16 + m) * KSTEP + half * 16];
            b2.q[0] = *(const u32x4*)(rp);  b2.q[1] = *(const u32x4*)(rp + 8);
            accR[sub]  = __builtin_amdgcn_wmma_f32_16x16x32_bf16(false, ax.v, false, b0.v, (short)0, accR[sub],  false, false);
            accZ[sub]  = __builtin_amdgcn_wmma_f32_16x16x32_bf16(false, ax.v, false, b1.v, (short)0, accZ[sub],  false, false);
            accXn[sub] = __builtin_amdgcn_wmma_f32_16x16x32_bf16(false, ax.v, false, b2.v, (short)0, accXn[sub], false, false);
            rp = &sB[((6 + sub) * 16 + m) * KSTEP + half * 16];
            b0.q[0] = *(const u32x4*)(rp);  b0.q[1] = *(const u32x4*)(rp + 8);
            rp = &sB[((8 + sub) * 16 + m) * KSTEP + half * 16];
            b1.q[0] = *(const u32x4*)(rp);  b1.q[1] = *(const u32x4*)(rp + 8);
            rp = &sB[((10 + sub) * 16 + m) * KSTEP + half * 16];
            b2.q[0] = *(const u32x4*)(rp);  b2.q[1] = *(const u32x4*)(rp + 8);
            accR[sub]  = __builtin_amdgcn_wmma_f32_16x16x32_bf16(false, ah.v, false, b0.v, (short)0, accR[sub],  false, false);
            accZ[sub]  = __builtin_amdgcn_wmma_f32_16x16x32_bf16(false, ah.v, false, b1.v, (short)0, accZ[sub],  false, false);
            accHn[sub] = __builtin_amdgcn_wmma_f32_16x16x32_bf16(false, ah.v, false, b2.v, (short)0, accHn[sub], false, false);
        }
    }

    const int row0 = btile * TILE_B + wave * 16 + half * 8;
    #pragma unroll
    for (int sub = 0; sub < 2; ++sub) {
        const int   col = ncol0 + sub * 16 + m;
        const float br_ = bih[u * G3 + col]         + bhh[u * G3 + col];
        const float bz_ = bih[u * G3 + H_DIM + col] + bhh[u * G3 + H_DIM + col];
        const float bxn = bih[u * G3 + 2 * H_DIM + col];
        const float bhn = bhh[u * G3 + 2 * H_DIM + col];
        #pragma unroll
        for (int j = 0; j < 8; ++j) {
            const size_t off = (size_t)(row0 + j) * (size_t)(U_DIM * H_DIM)
                             + (size_t)u * H_DIM + (size_t)col;
            const float hp = hidden[off];
            const float r  = fast_sigmoid(accR[sub][j] + br_);
            const float z  = fast_sigmoid(accZ[sub][j] + bz_);
            const float n  = fast_tanh(accXn[sub][j] + bxn + r * (accHn[sub][j] + bhn));
            out[off] = (1.0f - z) * n + z * hp;
        }
    }
}

extern "C" void kernel_launch(void* const* d_in, const int* in_sizes, int n_in,
                              void* d_out, int out_size, void* d_ws, size_t ws_size,
                              hipStream_t stream) {
    (void)in_sizes; (void)n_in; (void)out_size;

    const float* inputs = (const float*)d_in[0];
    const float* hidden = (const float*)d_in[1];
    const float* Wih    = (const float*)d_in[2];
    const float* Whh    = (const float*)d_in[3];
    const float* bih    = (const float*)d_in[4];
    const float* bhh    = (const float*)d_in[5];
    float*       out    = (float*)d_out;

    const size_t nW = (size_t)U_DIM * G3 * I_DIM;     // 6,291,456
    const size_t nX = (size_t)B_DIM * U_DIM * I_DIM;  // 4,194,304
    const size_t need = (2 * nW + 2 * nX) * sizeof(unsigned short);  // ~42 MB

    dim3 grid(H_DIM / TILE_N, B_DIM / TILE_B, U_DIM);   // (16, 8, 8)
    dim3 block(256);

    if (ws_size >= need) {
        unsigned short* wsWih = (unsigned short*)d_ws;
        unsigned short* wsWhh = wsWih + nW;
        unsigned short* wsX   = wsWhh + nW;
        unsigned short* wsH   = wsX + nX;

        const int w8 = (int)(nW / 8);
        const int x8 = (int)(nX / 8);
        cvt_f32_bf16_kernel<<<(w8 + 255) / 256, 256, 0, stream>>>(Wih, wsWih, w8);
        cvt_f32_bf16_kernel<<<(w8 + 255) / 256, 256, 0, stream>>>(Whh, wsWhh, w8);
        cvt_f32_bf16_kernel<<<(x8 + 255) / 256, 256, 0, stream>>>(inputs, wsX, x8);
        cvt_f32_bf16_kernel<<<(x8 + 255) / 256, 256, 0, stream>>>(hidden, wsH, x8);

        gru_wmma_bf16_pre<<<grid, block, 0, stream>>>(wsX, wsH, wsWih, wsWhh,
                                                      hidden, bih, bhh, out);
    } else {
        gru_wmma_bf16_fused<<<grid, block, 0, stream>>>(inputs, hidden, Wih, Whh,
                                                        bih, bhh, out);
    }
}
